// Wrap_Model_26044681683088
// MI455X (gfx1250) — matrix operand order repaired and checked by reading, not compiled, not run on hardware
//
#include <hip/hip_runtime.h>
#include <hip/hip_bf16.h>
#include <stdint.h>

typedef __bf16 bf16;
typedef __attribute__((ext_vector_type(16))) __bf16 bf16x16;
typedef __attribute__((ext_vector_type(8)))  __bf16 bf16x8;
typedef __attribute__((ext_vector_type(8)))  float  f32x8;
typedef __attribute__((ext_vector_type(4)))  float  f32x4;

#define N_TRAIN 200000
#define BQ      1024      // batch (rows of x / output)
#define DQ      512       // feature dim
#define INQ     1024      // input dim
#define NTILE   256       // train columns per block in distance kernels
#define NCHUNKS ((N_TRAIN + NTILE - 1) / NTILE)   // 782
#define BIGF    3.0e38f

__device__ __forceinline__ void top2_update(float& m1, float& m2, float x) {
    if (x < m1) { m2 = m1; m1 = x; }
    else if (x < m2) { m2 = x; }
}
__device__ __forceinline__ void top2_merge(float& m1, float& m2, float b1, float b2) {
    float lo = fminf(m1, b1);
    float hi = fminf(fmaxf(m1, b1), fminf(m2, b2));
    m1 = lo; m2 = hi;
}

// ---------------------------------------------------------------------------
// Kernel 0: W [1024x512] fp32 -> Wt [512x1024] bf16 (row n = column n of W)
// ---------------------------------------------------------------------------
__global__ void k_transposeW(const float* __restrict__ W, bf16* __restrict__ Wt) {
    int idx = blockIdx.x * blockDim.x + threadIdx.x;   // 0 .. 512*1024-1
    int n = idx >> 10;
    int k = idx & 1023;
    Wt[idx] = (bf16)W[(size_t)k * DQ + n];
}

// ---------------------------------------------------------------------------
// Kernel 1: feats = x @ W via bf16 WMMA.  Block tile 64x128, 8 waves (2Mx4N),
// wave tile 32x32 (2 M-subtiles x 2 N-subtiles). Stores feats as bf16.
// ---------------------------------------------------------------------------
__global__ void __launch_bounds__(256)
k_feats(const float* __restrict__ x, const bf16* __restrict__ Wt,
        bf16* __restrict__ featsB) {
    const int lane = threadIdx.x & 31;
    const int wid  = threadIdx.x >> 5;
    const int l15  = lane & 15;
    const int h    = lane >> 4;
    const int wm   = wid & 1;
    const int wn   = wid >> 1;
    const int mbase = blockIdx.x * 64  + wm * 32;
    const int nbase = blockIdx.y * 128 + wn * 32;

    f32x8 acc[2][2] = {};
    for (int kt = 0; kt < INQ; kt += 32) {
        bf16x16 a[2];
#pragma unroll
        for (int ms = 0; ms < 2; ++ms) {
            const float* ap = x + (size_t)(mbase + ms * 16 + l15) * INQ + kt + 16 * h;
            const f32x4* ap4 = (const f32x4*)ap;
            f32x4 r0 = ap4[0], r1 = ap4[1];     // K = kt+16h .. +7
            f32x4 r2 = ap4[2], r3 = ap4[3];     // K = kt+16h+8 .. +15
            bf16x16 av;
#pragma unroll
            for (int e = 0; e < 4; ++e) {
                av[e]      = (bf16)r0[e];
                av[4 + e]  = (bf16)r1[e];
                av[8 + e]  = (bf16)r2[e];
                av[12 + e] = (bf16)r3[e];
            }
            a[ms] = av;
        }
#pragma unroll
        for (int ns = 0; ns < 2; ++ns) {
            const bf16* bp = Wt + (size_t)(nbase + ns * 16 + l15) * INQ + kt + 16 * h;
            bf16x16 b = *(const bf16x16*)bp;
#pragma unroll
            for (int ms = 0; ms < 2; ++ms)
                acc[ms][ns] = __builtin_amdgcn_wmma_f32_16x16x32_bf16(
                    false, a[ms], false, b, (short)0, acc[ms][ns], false, false);
        }
    }
#pragma unroll
    for (int ms = 0; ms < 2; ++ms)
#pragma unroll
        for (int ns = 0; ns < 2; ++ns)
#pragma unroll
            for (int v = 0; v < 8; ++v) {
                int row = mbase + ms * 16 + 8 * h + v;
                int col = nbase + ns * 16 + l15;
                featsB[(size_t)row * DQ + col] = (bf16)acc[ms][ns][v];
            }
}

// ---------------------------------------------------------------------------
// Kernel 2: tsq[row] = sum_k featsB[row][k]^2  (fp32 accumulate)
// ---------------------------------------------------------------------------
__global__ void __launch_bounds__(256)
k_tsq(const bf16* __restrict__ featsB, float* __restrict__ tsq) {
    int row = blockIdx.x;
    int t   = threadIdx.x;
    float v0 = (float)featsB[(size_t)row * DQ + t];
    float v1 = (float)featsB[(size_t)row * DQ + t + 256];
    float s  = v0 * v0 + v1 * v1;
#pragma unroll
    for (int m = 1; m <= 16; m <<= 1) s += __shfl_xor(s, m, 32);
    __shared__ float red[8];
    if ((t & 31) == 0) red[t >> 5] = s;
    __syncthreads();
    if (t == 0) {
        float tot = 0.f;
#pragma unroll
        for (int i = 0; i < 8; ++i) tot += red[i];
        tsq[row] = tot;
    }
}

// ---------------------------------------------------------------------------
// Kernel 2b (big-ws path): one-shot fp32->bf16 conversion of train plus
// per-row r^2 in fp32.  One wave per train row: 32 lanes x 16 floats = 512.
// Memory-bound (reads 410 MB, writes 205 MB); removes ALL conversion VALU
// from the hot WMMA loop.
// ---------------------------------------------------------------------------
__global__ void __launch_bounds__(256)
k_cvt(const float* __restrict__ train, bf16* __restrict__ trainB,
      float* __restrict__ rsq) {
    const int wid  = threadIdx.x >> 5;
    const int lane = threadIdx.x & 31;
    const int row  = blockIdx.x * 8 + wid;            // 25000 blocks * 8 rows
    const float* src = train + (size_t)row * DQ + lane * 16;
    const f32x4* s4 = (const f32x4*)src;
    f32x4 t0 = s4[0], t1 = s4[1], t2 = s4[2], t3 = s4[3];
    bf16x16 b;
#pragma unroll
    for (int e = 0; e < 4; ++e) {
        b[e]      = (bf16)t0[e];
        b[4 + e]  = (bf16)t1[e];
        b[8 + e]  = (bf16)t2[e];
        b[12 + e] = (bf16)t3[e];
    }
    *(bf16x16*)(trainB + (size_t)row * DQ + lane * 16) = b;
    f32x4 sq = t0 * t0 + t1 * t1 + t2 * t2 + t3 * t3;
    float s = sq[0] + sq[1] + sq[2] + sq[3];
#pragma unroll
    for (int m = 1; m <= 16; m <<= 1) s += __shfl_xor(s, m, 32);
    if (lane == 0) rsq[row] = s;
}

// ---------------------------------------------------------------------------
// Kernel 3a (big-ws path): conversion-free fused distance GEMM + top-2.
// Inner loop per K-step: 4 bf16 A-frag loads, 4 bf16x16 B-frag loads
// (contiguous 32B K-run of a train row, same k = 16h+s order as A), 16 WMMAs.
// ---------------------------------------------------------------------------
__global__ void __launch_bounds__(256)
k_dist_pre(const bf16*  __restrict__ featsB, const bf16* __restrict__ trainB,
           const float* __restrict__ tsq,    const float* __restrict__ rsq,
           float2*      __restrict__ partial) {
    const int lane = threadIdx.x & 31;
    const int wid  = threadIdx.x >> 5;
    const int l15  = lane & 15;
    const int h    = lane >> 4;
    const int wm   = wid & 1;
    const int wn   = wid >> 1;
    const int mblk  = blockIdx.x;
    const int chunk = blockIdx.y;
    const int mbase = mblk * 128 + wm * 64;
    const int nbase = chunk * NTILE + wn * 64;

    f32x8 acc[4][4] = {};

    const bf16* brow[4];
    int jcl[4];
#pragma unroll
    for (int ns = 0; ns < 4; ++ns) {
        int j = nbase + ns * 16 + l15;
        if (j >= N_TRAIN) j = N_TRAIN - 1;           // clamp; masked to +inf later
        jcl[ns] = j;
        brow[ns] = trainB + (size_t)j * DQ + 16 * h;
    }
    const bf16* arow[4];
#pragma unroll
    for (int ms = 0; ms < 4; ++ms)
        arow[ms] = featsB + (size_t)(mbase + ms * 16 + l15) * DQ + 16 * h;

    for (int kt = 0; kt < DQ; kt += 32) {
        bf16x16 a[4];
#pragma unroll
        for (int ms = 0; ms < 4; ++ms) {
            bf16x8 lo = *(const bf16x8*)(arow[ms] + kt);        // K = kt+16h..+7
            bf16x8 hi = *(const bf16x8*)(arow[ms] + kt + 8);    // K = kt+16h+8..+15
            a[ms] = __builtin_shufflevector(lo, hi,
                     0,1,2,3,4,5,6,7,8,9,10,11,12,13,14,15);
        }
#pragma unroll
        for (int ns = 0; ns < 4; ++ns) {
            bf16x16 b = *(const bf16x16*)(brow[ns] + kt);       // K = kt+16h..+15
            __builtin_prefetch(brow[ns] + kt + 128, 0, 0);
#pragma unroll
            for (int ms = 0; ms < 4; ++ms)
                acc[ms][ns] = __builtin_amdgcn_wmma_f32_16x16x32_bf16(
                    false, a[ms], false, b, (short)0, acc[ms][ns], false, false);
        }
    }

    float rv[4];
#pragma unroll
    for (int ns = 0; ns < 4; ++ns) rv[ns] = rsq[jcl[ns]];

    __shared__ float2 lds_pairs[128][4];

#pragma unroll
    for (int ms = 0; ms < 4; ++ms) {
        const float* tp = tsq + mbase + ms * 16 + 8 * h;
        float tv[8];
#pragma unroll
        for (int v = 0; v < 8; ++v) tv[v] = tp[v];
#pragma unroll
        for (int v = 0; v < 8; ++v) {
            float m1 = BIGF, m2 = BIGF;
#pragma unroll
            for (int ns = 0; ns < 4; ++ns) {
                int j = nbase + ns * 16 + l15;
                float d2 = tv[v] + rv[ns] - 2.0f * acc[ms][ns][v];
                if (j >= N_TRAIN) d2 = BIGF;
                top2_update(m1, m2, d2);
            }
#pragma unroll
            for (int mask = 1; mask <= 8; mask <<= 1) {
                float o1 = __shfl_xor(m1, mask, 32);
                float o2 = __shfl_xor(m2, mask, 32);
                top2_merge(m1, m2, o1, o2);
            }
            if (l15 == 0) {
                int lr = wm * 64 + ms * 16 + 8 * h + v;
                lds_pairs[lr][wn] = make_float2(m1, m2);
            }
        }
    }
    __syncthreads();
    if (threadIdx.x < 128) {
        int lr = threadIdx.x;
        float2 p0 = lds_pairs[lr][0], p1 = lds_pairs[lr][1];
        float2 p2 = lds_pairs[lr][2], p3 = lds_pairs[lr][3];
        float m1 = p0.x, m2 = p0.y;
        top2_merge(m1, m2, p1.x, p1.y);
        top2_merge(m1, m2, p2.x, p2.y);
        top2_merge(m1, m2, p3.x, p3.y);
        int grow = mblk * 128 + lr;
        partial[(size_t)grow * NCHUNKS + chunk] = make_float2(m1, m2);
    }
}

// ---------------------------------------------------------------------------
// Kernel 3b (fallback, small ws): fused conversion + distance GEMM + top-2.
// ---------------------------------------------------------------------------
__global__ void __launch_bounds__(256)
k_dist(const bf16*  __restrict__ featsB, const float* __restrict__ train,
       const float* __restrict__ tsq,    float2*      __restrict__ partial) {
    const int lane = threadIdx.x & 31;
    const int wid  = threadIdx.x >> 5;
    const int l15  = lane & 15;
    const int h    = lane >> 4;
    const int wm   = wid & 1;
    const int wn   = wid >> 1;
    const int mblk  = blockIdx.x;
    const int chunk = blockIdx.y;
    const int mbase = mblk * 128 + wm * 64;
    const int nbase = chunk * NTILE + wn * 64;

    f32x8 acc[4][4] = {};
    float ssq[4] = {0.f, 0.f, 0.f, 0.f};

    const float* brow[4];
#pragma unroll
    for (int ns = 0; ns < 4; ++ns) {
        int j = nbase + ns * 16 + l15;
        if (j >= N_TRAIN) j = N_TRAIN - 1;
        brow[ns] = train + (size_t)j * DQ + 16 * h;
    }
    const bf16* arow[4];
#pragma unroll
    for (int ms = 0; ms < 4; ++ms)
        arow[ms] = featsB + (size_t)(mbase + ms * 16 + l15) * DQ + 16 * h;

    for (int kt = 0; kt < DQ; kt += 32) {
        bf16x16 a[4];
#pragma unroll
        for (int ms = 0; ms < 4; ++ms) {
            bf16x8 lo = *(const bf16x8*)(arow[ms] + kt);
            bf16x8 hi = *(const bf16x8*)(arow[ms] + kt + 8);
            a[ms] = __builtin_shufflevector(lo, hi,
                     0,1,2,3,4,5,6,7,8,9,10,11,12,13,14,15);
        }
#pragma unroll
        for (int ns = 0; ns < 4; ++ns) {
            const f32x4* bp = (const f32x4*)(brow[ns] + kt);
            f32x4 t0 = bp[0], t1 = bp[1], t2 = bp[2], t3 = bp[3];
            __builtin_prefetch(brow[ns] + kt + 128, 0, 0);
            f32x4 sq = t0 * t0 + t1 * t1 + t2 * t2 + t3 * t3;
            ssq[ns] += sq[0] + sq[1] + sq[2] + sq[3];
            bf16x16 b;
#pragma unroll
            for (int e = 0; e < 4; ++e) {
                b[e]      = (bf16)t0[e];
                b[4 + e]  = (bf16)t1[e];
                b[8 + e]  = (bf16)t2[e];
                b[12 + e] = (bf16)t3[e];
            }
#pragma unroll
            for (int ms = 0; ms < 4; ++ms)
                acc[ms][ns] = __builtin_amdgcn_wmma_f32_16x16x32_bf16(
                    false, a[ms], false, b, (short)0, acc[ms][ns], false, false);
        }
    }

    float rsq[4];
#pragma unroll
    for (int ns = 0; ns < 4; ++ns)
        rsq[ns] = ssq[ns] + __shfl_xor(ssq[ns], 16, 32);

    __shared__ float2 lds_pairs[128][4];

#pragma unroll
    for (int ms = 0; ms < 4; ++ms) {
        const float* tp = tsq + mbase + ms * 16 + 8 * h;
        float tv[8];
#pragma unroll
        for (int v = 0; v < 8; ++v) tv[v] = tp[v];
#pragma unroll
        for (int v = 0; v < 8; ++v) {
            float m1 = BIGF, m2 = BIGF;
#pragma unroll
            for (int ns = 0; ns < 4; ++ns) {
                int j = nbase + ns * 16 + l15;
                float d2 = tv[v] + rsq[ns] - 2.0f * acc[ms][ns][v];
                if (j >= N_TRAIN) d2 = BIGF;
                top2_update(m1, m2, d2);
            }
#pragma unroll
            for (int mask = 1; mask <= 8; mask <<= 1) {
                float o1 = __shfl_xor(m1, mask, 32);
                float o2 = __shfl_xor(m2, mask, 32);
                top2_merge(m1, m2, o1, o2);
            }
            if (l15 == 0) {
                int lr = wm * 64 + ms * 16 + 8 * h + v;
                lds_pairs[lr][wn] = make_float2(m1, m2);
            }
        }
    }
    __syncthreads();
    if (threadIdx.x < 128) {
        int lr = threadIdx.x;
        float2 p0 = lds_pairs[lr][0], p1 = lds_pairs[lr][1];
        float2 p2 = lds_pairs[lr][2], p3 = lds_pairs[lr][3];
        float m1 = p0.x, m2 = p0.y;
        top2_merge(m1, m2, p1.x, p1.y);
        top2_merge(m1, m2, p2.x, p2.y);
        top2_merge(m1, m2, p3.x, p3.y);
        int grow = mblk * 128 + lr;
        partial[(size_t)grow * NCHUNKS + chunk] = make_float2(m1, m2);
    }
}

// ---------------------------------------------------------------------------
// Kernel 4: per-row merge of NCHUNKS partial top-2 pairs -> out = m1+m2
// ---------------------------------------------------------------------------
__global__ void __launch_bounds__(256)
k_final(const float2* __restrict__ partial, float* __restrict__ out) {
    int row = blockIdx.x;
    int t   = threadIdx.x;
    float m1 = BIGF, m2 = BIGF;
    for (int c = t; c < NCHUNKS; c += 256) {
        float2 p = partial[(size_t)row * NCHUNKS + c];
        top2_merge(m1, m2, p.x, p.y);
    }
#pragma unroll
    for (int mask = 1; mask <= 16; mask <<= 1) {
        float o1 = __shfl_xor(m1, mask, 32);
        float o2 = __shfl_xor(m2, mask, 32);
        top2_merge(m1, m2, o1, o2);
    }
    __shared__ float2 red[8];
    if ((t & 31) == 0) red[t >> 5] = make_float2(m1, m2);
    __syncthreads();
    if (t == 0) {
        float a1 = red[0].x, a2 = red[0].y;
#pragma unroll
        for (int i = 1; i < 8; ++i) top2_merge(a1, a2, red[i].x, red[i].y);
        out[row] = a1 + a2;
    }
}

// ---------------------------------------------------------------------------
extern "C" void kernel_launch(void* const* d_in, const int* in_sizes, int n_in,
                              void* d_out, int out_size, void* d_ws, size_t ws_size,
                              hipStream_t stream) {
    (void)in_sizes; (void)n_in; (void)out_size;
    const float* x     = (const float*)d_in[0];   // [1024,1024]
    const float* W     = (const float*)d_in[1];   // [1024,512]
    const float* train = (const float*)d_in[2];   // [200000,512]
    float* out = (float*)d_out;                   // [1024]

    uint8_t* ws = (uint8_t*)d_ws;
    bf16*   featsB  = (bf16*)ws;                                   // 1 MB
    bf16*   Wt      = (bf16*)(ws + (1u << 20));                    // 1 MB
    float*  tsq     = (float*)(ws + (2u << 20));                   // 4 KB
    float2* partial = (float2*)(ws + (2u << 20) + (1u << 16));     // ~6.4 MB
    // big-ws-only buffers
    float*  rsq     = (float*)(ws + (16u << 20));                  // 800 KB
    bf16*   trainB  = (bf16*)(ws + (17u << 20));                   // 204.8 MB
    const size_t need_big = (size_t)(17u << 20) + (size_t)N_TRAIN * DQ * 2;

    k_transposeW<<<dim3((DQ * INQ) / 256), 256, 0, stream>>>(W, Wt);
    k_feats<<<dim3(BQ / 64, DQ / 128), 256, 0, stream>>>(x, Wt, featsB);
    k_tsq<<<dim3(BQ), 256, 0, stream>>>(featsB, tsq);

    if (ws_size >= need_big) {
        // preferred path: one-shot conversion, conversion-free WMMA loop
        k_cvt<<<dim3(N_TRAIN / 8), 256, 0, stream>>>(train, trainB, rsq);
        k_dist_pre<<<dim3(BQ / 128, NCHUNKS), 256, 0, stream>>>(
            featsB, trainB, tsq, rsq, partial);
    } else {
        // fallback: fused conversion inside the GEMM
        k_dist<<<dim3(BQ / 128, NCHUNKS), 256, 0, stream>>>(
            featsB, train, tsq, partial);
    }
    k_final<<<dim3(BQ), 256, 0, stream>>>(partial, out);
}
